// Block_39625368273507
// MI455X (gfx1250) — compile-verified, matrix operand order checked by reading, not run on hardware
//
#include <hip/hip_runtime.h>
#include <hip/hip_bf16.h>
#include <math.h>

// ---------- types ----------
typedef __attribute__((ext_vector_type(16))) __bf16 v16bf;
typedef __attribute__((ext_vector_type(8)))  __bf16 v8bf;
typedef __attribute__((ext_vector_type(8)))  float  v8f;
typedef __attribute__((ext_vector_type(4)))  int    v4i;

#define DEV static __device__ __forceinline__

// ---------- gfx1250 async global->LDS copy (guarded; falls back to sync) ----------
#if defined(__AMDGCN__) && __has_builtin(__builtin_amdgcn_global_load_async_to_lds_b128)
#define USE_ASYNC_LDS 1
#else
#define USE_ASYNC_LDS 0
#endif

DEV void cp16_to_lds(const __bf16* g, __bf16* l) {
#if USE_ASYNC_LDS
  __builtin_amdgcn_global_load_async_to_lds_b128(
      (__attribute__((address_space(1))) v4i*)g,
      (__attribute__((address_space(3))) v4i*)l,
      0, 0);
#else
  *(v8bf*)l = *(const v8bf*)g;
#endif
}

DEV void wait_async_lds() {
#if USE_ASYNC_LDS
#if __has_builtin(__builtin_amdgcn_s_wait_asynccnt)
  __builtin_amdgcn_s_wait_asynccnt(0);
#else
  asm volatile("s_wait_asynccnt 0x0" ::: "memory");
#endif
#endif
}

// fp32 -> bf16 round-to-nearest-even
DEV __bf16 f2bf(float f) {
  union { float f; unsigned u; } a; a.f = f;
  unsigned r = (a.u + 0x7FFFu + ((a.u >> 16) & 1u)) >> 16;
  union { unsigned short s; __bf16 b; } o; o.s = (unsigned short)r;
  return o.b;
}

// Load a 16x32 bf16 A/B fragment per ISA layout:
// lane holds K = [lk, lk+8) in elems 0..7 and K = [lk+16, lk+24) in elems 8..15,
// where caller points p at (row_base + lk). lk = 8*(lane>=16).
DEV v16bf ldfrag(const __bf16* p) {
  union { v16bf v; v8bf h[2]; } u;
  u.h[0] = *(const v8bf*)p;
  u.h[1] = *(const v8bf*)(p + 16);
  return u.v;
}

DEV v8f wmma_bf16(v16bf a, v16bf b, v8f c) {
  return __builtin_amdgcn_wmma_f32_16x16x32_bf16(false, a, false, b, (short)0, c, false, false);
}

// ---------- problem constants ----------
#define BD   768
#define NHID 3072
#define NB   8
#define NN   1024
#define NH   12
#define HD   64
#define MROWS (NB*NN)        // 8192
#define S1C  320
#define S2C  384

// ---------- fp32 -> bf16 conversion ----------
__global__ void cvt_bf16_kernel(const float* __restrict__ in, __bf16* __restrict__ out, int n) {
  int i = blockIdx.x * blockDim.x + threadIdx.x;
  if (i < n) out[i] = f2bf(in[i]);
}

// ---------- split LayerNorm: fp32 [8192,768] -> bf16 [8192,768] ----------
__global__ __launch_bounds__(128) void ln_split_kernel(
    const float* __restrict__ x,
    const float* __restrict__ ga, const float* __restrict__ ba,
    const float* __restrict__ gb, const float* __restrict__ bb,
    const float* __restrict__ gc, const float* __restrict__ bc,
    __bf16* __restrict__ out)
{
  const int row = blockIdx.x;
  const int t = threadIdx.x;
  const float* xr = x + (size_t)row * BD;

  float v[6];
  float s[3] = {0.f, 0.f, 0.f};
  float q[3] = {0.f, 0.f, 0.f};
#pragma unroll
  for (int i = 0; i < 6; ++i) {
    int c = t + i * 128;
    float xv = xr[c];
    v[i] = xv;
    int g = (c < S1C) ? 0 : ((c < S2C) ? 1 : 2);
    s[g] += xv; q[g] += xv * xv;
  }

  __shared__ float rs[3][128], rq[3][128];
#pragma unroll
  for (int g = 0; g < 3; ++g) { rs[g][t] = s[g]; rq[g][t] = q[g]; }
  __syncthreads();
  for (int st = 64; st > 0; st >>= 1) {
    if (t < st) {
#pragma unroll
      for (int g = 0; g < 3; ++g) { rs[g][t] += rs[g][t + st]; rq[g][t] += rq[g][t + st]; }
    }
    __syncthreads();
  }
  __shared__ float mean[3], rstd[3];
  if (t == 0) {
    const float cnt[3] = {320.f, 64.f, 384.f};
#pragma unroll
    for (int g = 0; g < 3; ++g) {
      float m = rs[g][0] / cnt[g];
      float var = rq[g][0] / cnt[g] - m * m;
      mean[g] = m; rstd[g] = rsqrtf(var + 1e-5f);
    }
  }
  __syncthreads();
#pragma unroll
  for (int i = 0; i < 6; ++i) {
    int c = t + i * 128;
    int g = (c < S1C) ? 0 : ((c < S2C) ? 1 : 2);
    float gm, bt;
    if (g == 0)      { gm = ga[c];        bt = ba[c]; }
    else if (g == 1) { gm = gb[c - S1C];  bt = bb[c - S1C]; }
    else             { gm = gc[c - S2C];  bt = bc[c - S2C]; }
    out[(size_t)row * BD + c] = f2bf((v[i] - mean[g]) * rstd[g] * gm + bt);
  }
}

// ---------- WMMA GEMM: C[M,N] = A[M,K] * W[N,K]^T, templated epilogue ----------
// Double-buffered LDS staging via async global->LDS copies (ASYNCcnt) when
// available; one barrier per K-step. MODEs:
// 0: QKV scatter -> q[B,H,N,64], k[B,H,N,64], vt[B,H,64,N] (bf16)
// 1: proj: outf = acc + bias + resid   (fp32)
// 2: fc1:  outb = gelu(acc + bias)     (bf16)
// 3: fc2:  outf = acc + bias + resid   (fp32)
template<int MODE>
__global__ __launch_bounds__(128) void gemm_bf16_kernel(
    const __bf16* __restrict__ A, const __bf16* __restrict__ W,
    const float* __restrict__ bias, const float* __restrict__ resid,
    float* __restrict__ outf, __bf16* __restrict__ outb,
    __bf16* __restrict__ qo, __bf16* __restrict__ ko, __bf16* __restrict__ vto,
    int M, int N, int K)
{
  constexpr int LDT = 40;                      // 32 K-halves + 8 pad
  __shared__ __bf16 As[2][64 * LDT];
  __shared__ __bf16 Bs[2][64 * LDT];

  const int t = threadIdx.x;
  const int lane = t & 31;
  const int w = t >> 5;
  const int bm0 = blockIdx.y * 64;
  const int bn0 = blockIdx.x * 64;
  const int lr = lane & 15;
  const int lk = (lane >> 4) * 8;
  const int mo = (lane >> 4) * 8;

  // per-thread staging job: 2x16 halves of one row in each tile
  const int r = t >> 1;
  const int p = (t & 1) * 16;
  const __bf16* gA = A + (size_t)(bm0 + r) * K + p;
  const __bf16* gW = W + (size_t)(bn0 + r) * K + p;

  auto issue = [&](int buf, int k0) {
    cp16_to_lds(gA + k0,     &As[buf][r * LDT + p]);
    cp16_to_lds(gA + k0 + 8, &As[buf][r * LDT + p + 8]);
    cp16_to_lds(gW + k0,     &Bs[buf][r * LDT + p]);
    cp16_to_lds(gW + k0 + 8, &Bs[buf][r * LDT + p + 8]);
  };

  const v8f z = {0.f,0.f,0.f,0.f,0.f,0.f,0.f,0.f};
  v8f acc[4] = {z, z, z, z};

  issue(0, 0);
  int buf = 0;
  for (int k0 = 0; k0 < K; k0 += 32) {
    wait_async_lds();          // own async copies into `buf` done
    __syncthreads();           // everyone's copies visible; prev reads finished
    if (k0 + 32 < K) issue(buf ^ 1, k0 + 32);

    v16bf a = ldfrag(&As[buf][(w * 16 + lr) * LDT + lk]);
#pragma unroll
    for (int u = 0; u < 4; ++u) {
      v16bf b = ldfrag(&Bs[buf][(u * 16 + lr) * LDT + lk]);
      acc[u] = wmma_bf16(a, b, acc[u]);
    }
    buf ^= 1;
  }

  // epilogue: C layout — VGPR j holds (M = j + 8*(lane>=16), N = lane&15)
#pragma unroll
  for (int u = 0; u < 4; ++u) {
#pragma unroll
    for (int j = 0; j < 8; ++j) {
      int row = bm0 + w * 16 + mo + j;
      int col = bn0 + u * 16 + lr;
      float val = acc[u][j];
      if constexpr (MODE == 0) {
        int s = col / BD, rem = col - s * BD;
        int h = rem >> 6, e = rem & 63;
        int bb_ = row >> 10, n = row & 1023;
        size_t bh = (size_t)(bb_ * NH + h);
        __bf16 bv = f2bf(val);
        if (s == 0)      qo[(bh * NN + n) * HD + e] = bv;
        else if (s == 1) ko[(bh * NN + n) * HD + e] = bv;
        else             vto[(bh * HD + e) * NN + n] = bv;
      } else if constexpr (MODE == 2) {
        float g = val + bias[col];
        g = 0.5f * g * (1.0f + erff(g * 0.70710678118654752f));
        outb[(size_t)row * N + col] = f2bf(g);
      } else {
        size_t o = (size_t)row * BD + col;
        outf[o] = val + bias[col] + resid[o];
      }
    }
  }
}

// ---------- attention: softmax(Q K^T * 0.125) V, one wave per (b,h,16 rows) ----------
__global__ __launch_bounds__(32) void attn_kernel(
    const __bf16* __restrict__ q, const __bf16* __restrict__ k,
    const __bf16* __restrict__ vt, __bf16* __restrict__ y)
{
  __shared__ float sc[16 * NN];                // 64 KB score strip
  const int lane = threadIdx.x;
  const int lr = lane & 15;
  const int lk = (lane >> 4) * 8;
  const int mo = (lane >> 4) * 8;
  const int m0 = blockIdx.x * 16;
  const int h = blockIdx.y, b = blockIdx.z;
  const size_t bh = (size_t)b * NH + h;
  const __bf16* qb = q  + bh * (size_t)NN * HD;
  const __bf16* kb = k  + bh * (size_t)NN * HD;
  const __bf16* vb = vt + bh * (size_t)HD * NN;

  // Q fragments (16x64 = two 16x32 K-steps), kept in registers
  v16bf aq0 = ldfrag(qb + (size_t)(m0 + lr) * HD + lk);
  v16bf aq1 = ldfrag(qb + (size_t)(m0 + lr) * HD + 32 + lk);

  // scores = Q K^T * scale  -> LDS fp32
  for (int n0 = 0; n0 < NN; n0 += 16) {
    v16bf b0 = ldfrag(kb + (size_t)(n0 + lr) * HD + lk);
    v16bf b1 = ldfrag(kb + (size_t)(n0 + lr) * HD + 32 + lk);
    v8f c = {0.f,0.f,0.f,0.f,0.f,0.f,0.f,0.f};
    c = wmma_bf16(aq0, b0, c);
    c = wmma_bf16(aq1, b1, c);
#pragma unroll
    for (int j = 0; j < 8; ++j)
      sc[(mo + j) * NN + n0 + lr] = c[j] * 0.125f;
  }
  __syncthreads();

  // softmax stats: lane owns half of row lr; combine halves via shfl_xor(16)
  const float* srow = sc + lr * NN;
  const int c0 = (lane >> 4) * 512;
  float mx = -3.0e38f;
  for (int i = 0; i < 512; ++i) mx = fmaxf(mx, srow[c0 + i]);
  mx = fmaxf(mx, __shfl_xor(mx, 16, 32));
  float sum = 0.f;
  for (int i = 0; i < 512; ++i) sum += __expf(srow[c0 + i] - mx);
  sum += __shfl_xor(sum, 16, 32);
  const float rinv = 1.0f / sum;
  __syncthreads();

  // y_tile[16,64] = P[16,1024] * V[1024,64]; regen bf16 probs on the fly
  // (A-frag lane lr only ever needs row lr, whose mx/rinv this lane holds)
  const v8f z = {0.f,0.f,0.f,0.f,0.f,0.f,0.f,0.f};
  v8f acc[4] = {z, z, z, z};
  for (int k0 = 0; k0 < NN; k0 += 32) {
    v16bf a;
#pragma unroll
    for (int i = 0; i < 8; ++i) {
      a[i]     = f2bf(__expf(srow[k0 + lk + i]      - mx) * rinv);
      a[i + 8] = f2bf(__expf(srow[k0 + lk + 16 + i] - mx) * rinv);
    }
#pragma unroll
    for (int u = 0; u < 4; ++u) {
      v16bf bf = ldfrag(vb + (size_t)(u * 16 + lr) * NN + k0 + lk);
      acc[u] = wmma_bf16(a, bf, acc[u]);
    }
  }

  const size_t rb = (size_t)b * NN;
#pragma unroll
  for (int u = 0; u < 4; ++u)
#pragma unroll
    for (int j = 0; j < 8; ++j)
      y[(rb + m0 + mo + j) * BD + h * HD + u * 16 + lr] = f2bf(acc[u][j]);
}

// ---------- launch ----------
extern "C" void kernel_launch(void* const* d_in, const int* in_sizes, int n_in,
                              void* d_out, int out_size, void* d_ws, size_t ws_size,
                              hipStream_t stream) {
  (void)in_sizes; (void)n_in; (void)out_size; (void)ws_size;

  const float* x      = (const float*)d_in[0];
  const float* ln1a_g = (const float*)d_in[1];
  const float* ln1a_b = (const float*)d_in[2];
  const float* ln1b_g = (const float*)d_in[3];
  const float* ln1b_b = (const float*)d_in[4];
  const float* ln1c_g = (const float*)d_in[5];
  const float* ln1c_b = (const float*)d_in[6];
  const float* ln2a_g = (const float*)d_in[7];
  const float* ln2a_b = (const float*)d_in[8];
  const float* ln2b_g = (const float*)d_in[9];
  const float* ln2b_b = (const float*)d_in[10];
  const float* ln2c_g = (const float*)d_in[11];
  const float* ln2c_b = (const float*)d_in[12];
  const float* qkv_w  = (const float*)d_in[13];
  const float* proj_w = (const float*)d_in[14];
  const float* proj_b = (const float*)d_in[15];
  const float* fc1_w  = (const float*)d_in[16];
  const float* fc1_b  = (const float*)d_in[17];
  const float* fc2_w  = (const float*)d_in[18];
  const float* fc2_b  = (const float*)d_in[19];
  float* out = (float*)d_out;

  char* ws = (char*)d_ws;
  size_t off = 0;
  auto alloc = [&](size_t bytes) -> void* {
    off = (off + 255) & ~(size_t)255;
    void* p = ws + off;
    off += bytes;
    return p;
  };

  const size_t nQKVW = (size_t)3 * BD * BD;     // 2304*768
  const size_t nPRJW = (size_t)BD * BD;
  const size_t nFC1W = (size_t)NHID * BD;
  const size_t nFC2W = (size_t)BD * NHID;
  const size_t nTOK  = (size_t)MROWS * BD;      // 8192*768
  const size_t nHEADS = (size_t)NB * NH * NN * HD;

  __bf16* wqkv = (__bf16*)alloc(nQKVW * 2);
  __bf16* wprj = (__bf16*)alloc(nPRJW * 2);
  __bf16* wfc1 = (__bf16*)alloc(nFC1W * 2);
  __bf16* wfc2 = (__bf16*)alloc(nFC2W * 2);
  __bf16* xn   = (__bf16*)alloc(nTOK * 2);      // reused for LN1 and LN2 output
  __bf16* qb   = (__bf16*)alloc(nHEADS * 2);
  __bf16* kb   = (__bf16*)alloc(nHEADS * 2);
  __bf16* vtb  = (__bf16*)alloc(nHEADS * 2);
  __bf16* yb   = (__bf16*)alloc(nTOK * 2);
  float*  x1   = (float*)alloc(nTOK * 4);
  __bf16* hb   = (__bf16*)alloc((size_t)MROWS * NHID * 2);

  // 1) weights fp32 -> bf16
  cvt_bf16_kernel<<<(int)((nQKVW + 255) / 256), 256, 0, stream>>>(qkv_w, wqkv, (int)nQKVW);
  cvt_bf16_kernel<<<(int)((nPRJW + 255) / 256), 256, 0, stream>>>(proj_w, wprj, (int)nPRJW);
  cvt_bf16_kernel<<<(int)((nFC1W + 255) / 256), 256, 0, stream>>>(fc1_w, wfc1, (int)nFC1W);
  cvt_bf16_kernel<<<(int)((nFC2W + 255) / 256), 256, 0, stream>>>(fc2_w, wfc2, (int)nFC2W);

  // 2) split LN1
  ln_split_kernel<<<MROWS, 128, 0, stream>>>(x, ln1a_g, ln1a_b, ln1b_g, ln1b_b, ln1c_g, ln1c_b, xn);

  // 3) QKV GEMM (scatter into q/k/v^T)
  gemm_bf16_kernel<0><<<dim3(3 * BD / 64, MROWS / 64), 128, 0, stream>>>(
      xn, wqkv, nullptr, nullptr, nullptr, nullptr, qb, kb, vtb, MROWS, 3 * BD, BD);

  // 4) attention
  attn_kernel<<<dim3(NN / 16, NH, NB), 32, 0, stream>>>(qb, kb, vtb, yb);

  // 5) proj + bias + residual -> x1 (fp32)
  gemm_bf16_kernel<1><<<dim3(BD / 64, MROWS / 64), 128, 0, stream>>>(
      yb, wprj, proj_b, x, x1, nullptr, nullptr, nullptr, nullptr, MROWS, BD, BD);

  // 6) split LN2
  ln_split_kernel<<<MROWS, 128, 0, stream>>>(x1, ln2a_g, ln2a_b, ln2b_g, ln2b_b, ln2c_g, ln2c_b, xn);

  // 7) FC1 + bias + exact GELU -> hb (bf16)
  gemm_bf16_kernel<2><<<dim3(NHID / 64, MROWS / 64), 128, 0, stream>>>(
      xn, wfc1, fc1_b, nullptr, nullptr, hb, nullptr, nullptr, nullptr, MROWS, NHID, BD);

  // 8) FC2 + bias + residual -> out (fp32)
  gemm_bf16_kernel<3><<<dim3(BD / 64, MROWS / 64), 128, 0, stream>>>(
      hb, wfc2, fc2_b, x1, out, nullptr, nullptr, nullptr, nullptr, MROWS, BD, NHID);
}
